// Monotonic_Alignment_Search_32263794327645
// MI455X (gfx1250) — matrix operand order repaired and checked by reading, not compiled, not run on hardware
//
#include <hip/hip_runtime.h>
#include <hip/hip_bf16.h>

// ---------------------------------------------------------------------------
// Monotonic Alignment Search pipeline for MI455X (gfx1250, wave32, WMMA).
//
// Shapes: B=32, ENC_D=192, COND_D=192, FEAT_D=128, ENC_T=512, FEAT_T=2048, H=2
//
// All GEMMs use v_wmma_f32_16x16x32_f16.  LDS tiles are stored directly in
// the WMMA fragment layout ([subtile][lane][16 halves], 32B per lane), so a
// fragment load is one aligned v16h read (2x ds_load_b128) and staging is
// one ds_store_b128 per 8-half chunk.  nc2+nc3 are fused into a single K=256
// GEMM with the nc1+nc4 vector and band mask fused into the epilogue.  The
// Viterbi forward scan runs in place on neg_cent; backtrace is 1 lane/batch.
// ---------------------------------------------------------------------------

#define BATCH   32
#define ENC_D   192
#define FEAT_D  128
#define ENC_T   512
#define FEAT_T  2048
#define HEADS   2
#define DH      96          // ENC_D / HEADS
#define PROJ_D  256         // 2*FEAT_D
#define NEGC    (-1000000000.0f)

typedef __attribute__((ext_vector_type(16))) _Float16 v16h;
typedef __attribute__((ext_vector_type(8)))  _Float16 v8h;
typedef __attribute__((ext_vector_type(8)))  float    v8f;

// Epilogue mode bits for the generic GEMM
#define MODE_BIAS    1   // += bias[m]
#define MODE_RESID   2   // += resid[z][m][n]
#define MODE_COLMASK 4   // *= (n < enc_len[z])
#define MODE_NEGCENT 8   // += cvec[z][n]; *= (n < enc_len[z] && m < feat_len[z])

#define TILE_M 64
#define TILE_N 128
#define TILE_K 32

// ---------------------------------------------------------------------------
// Generic batched WMMA GEMM: C[z] = A[z] * B[z] (+ epilogue)
//   A: M x K  (row-major, lda; transA -> element (m,k) at A[k*lda + m])
//   B: K x N  (row-major, ldb; transB -> element (k,n) at B[n*ldb + k])
//   C: M x N  (row-major, ldc), f32
// Block: 128 threads = 4 waves. Tile 64x128, K-step 32.
// Wave w owns m-subtile w (16 rows) and iterates 8 n-subtiles (8 v8f accs).
//
// LDS fragment layout (matches ISA 16-bit A / B WMMA register layouts):
//   A 16x32: lane<16 holds row M=lane, K {0..7,16..23}; lane>=16 same row,
//            K {8..15,24..31}.  K-octet o -> (laneHi = o&1, halfBase=(o>>1)*8)
//   B 32x16: lane<16 holds col N=lane, K {0..7,16..23}; lane>=16 K {8..15,24..31}.
// ---------------------------------------------------------------------------
__global__ __launch_bounds__(128)
void gemm_wmma_f16(const float* __restrict__ A, long sA, int lda, int transA,
                   const float* __restrict__ Bm, long sB, int ldb, int transB,
                   float* __restrict__ C, long sC, int ldc,
                   int M, int N, int K,
                   const float* __restrict__ bias,
                   const float* __restrict__ resid, long sR, int ldr,
                   const float* __restrict__ cvec,
                   const int* __restrict__ enc_len,
                   const int* __restrict__ feat_len,
                   int mode)
{
  __shared__ __align__(32) _Float16 Af[4][32][16];   // 4 KB
  __shared__ __align__(32) _Float16 Bf[8][32][16];   // 8 KB

  const int z = blockIdx.z;
  const float* Ab = A  + (long)z * sA;
  const float* Bb = Bm + (long)z * sB;
  float*       Cb = C  + (long)z * sC;

  const int m0   = blockIdx.y * TILE_M;
  const int n0   = blockIdx.x * TILE_N;
  const int tid  = threadIdx.x;
  const int wave = tid >> 5;
  const int lane = tid & 31;

  v8f acc[8] = {};

  for (int kk = 0; kk < K; kk += TILE_K) {
    // ---- Stage A tile (64x32) into fragment layout: 256 x 16B chunks ----
    for (int c = tid; c < 256; c += 128) {
      const int seg   = c & 1;            // half-base = seg*8
      const int lane2 = (c >> 1) & 31;
      const int ms    = c >> 6;           // m-subtile 0..3
      const int row   = (ms << 4) + (lane2 & 15);
      const int oct   = (seg << 1) | (lane2 >> 4);
      const int ks    = kk + (oct << 3);
      const int gm    = m0 + row;
      v8h pk;
      if (!transA && gm < M && (ks + 7) < K) {
        const float4 f0 = *reinterpret_cast<const float4*>(Ab + (long)gm * lda + ks);
        const float4 f1 = *reinterpret_cast<const float4*>(Ab + (long)gm * lda + ks + 4);
        pk[0] = (_Float16)f0.x; pk[1] = (_Float16)f0.y;
        pk[2] = (_Float16)f0.z; pk[3] = (_Float16)f0.w;
        pk[4] = (_Float16)f1.x; pk[5] = (_Float16)f1.y;
        pk[6] = (_Float16)f1.z; pk[7] = (_Float16)f1.w;
      } else {
#pragma unroll
        for (int h = 0; h < 8; ++h) {
          const int gk = ks + h;
          float v = 0.0f;
          if (gm < M && gk < K)
            v = transA ? Ab[(long)gk * lda + gm] : Ab[(long)gm * lda + gk];
          pk[h] = (_Float16)v;
        }
      }
      *reinterpret_cast<v8h*>(&Af[ms][lane2][seg << 3]) = pk;
    }

    // ---- Stage B tile (32x128) into fragment layout: 512 x 16B chunks ----
    for (int c = tid; c < 512; c += 128) {
      const int seg   = c & 1;
      const int lane2 = (c >> 1) & 31;
      const int ns    = (c >> 6) & 7;     // n-subtile 0..7
      const int col   = (ns << 4) + (lane2 & 15);
      const int kb    = (seg << 4) | ((lane2 >> 4) << 3);
      const int ks    = kk + kb;
      const int gn    = n0 + col;
      v8h pk;
      if (transB && gn < N && (ks + 7) < K) {
        const float4 f0 = *reinterpret_cast<const float4*>(Bb + (long)gn * ldb + ks);
        const float4 f1 = *reinterpret_cast<const float4*>(Bb + (long)gn * ldb + ks + 4);
        pk[0] = (_Float16)f0.x; pk[1] = (_Float16)f0.y;
        pk[2] = (_Float16)f0.z; pk[3] = (_Float16)f0.w;
        pk[4] = (_Float16)f1.x; pk[5] = (_Float16)f1.y;
        pk[6] = (_Float16)f1.z; pk[7] = (_Float16)f1.w;
      } else {
#pragma unroll
        for (int h = 0; h < 8; ++h) {
          const int gk = ks + h;
          float v = 0.0f;
          if (gn < N && gk < K)
            v = transB ? Bb[(long)gn * ldb + gk] : Bb[(long)gk * ldb + gn];
          pk[h] = (_Float16)v;
        }
      }
      *reinterpret_cast<v8h*>(&Bf[ns][lane2][seg << 3]) = pk;
    }

    // Prefetch hint for the next K tile (emits global_prefetch_b8)
    if (kk + TILE_K < K && !transA)
      __builtin_prefetch(Ab + (long)(m0 + (tid >> 1)) * lda + kk + TILE_K, 0, 1);
    __syncthreads();

    // ---- Compute: 1 A-fragment, 8 WMMAs per wave per K-step ----
    const v16h af = *reinterpret_cast<const v16h*>(Af[wave][lane]);
#pragma unroll
    for (int nn = 0; nn < 8; ++nn) {
      const v16h bf = *reinterpret_cast<const v16h*>(Bf[nn][lane]);
      acc[nn] = __builtin_amdgcn_wmma_f32_16x16x32_f16(
          false, af, false, bf, (short)0, acc[nn], false, false);
    }
    __syncthreads();
  }

  // Epilogue: C/D layout -> lane<16: (M=r, N=lane); lane>=16: (M=8+r, N=lane-16)
  int elenB = 0, flenB = 0;
  if (mode & (MODE_COLMASK | MODE_NEGCENT)) elenB = enc_len[z];
  if (mode & MODE_NEGCENT) flenB = feat_len[z];

#pragma unroll
  for (int nn = 0; nn < 8; ++nn) {
#pragma unroll
    for (int r = 0; r < 8; ++r) {
      const int m = m0 + (wave << 4) + r + ((lane < 16) ? 0 : 8);
      const int n = n0 + (nn << 4) + (lane & 15);
      if (m < M && n < N) {
        float v = acc[nn][r];
        if (mode & MODE_BIAS)    v += bias[m];
        if (mode & MODE_RESID)   v += resid[(long)z * sR + (long)m * ldr + n];
        if (mode & MODE_NEGCENT) v += cvec[(long)z * ENC_T + n];
        if (mode & MODE_COLMASK) v  = (n < elenB) ? v : 0.0f;
        if (mode & MODE_NEGCENT) v  = ((n < elenB) && (m < flenB)) ? v : 0.0f;
        Cb[(long)m * ldc + n] = v;
      }
    }
  }
}

// ---------------------------------------------------------------------------
// Softmax over the channel dim (96 per head), one thread per (b,h,t) column.
// ---------------------------------------------------------------------------
__global__ void softmax_channel(float* __restrict__ q)
{
  const int idx = blockIdx.x * blockDim.x + threadIdx.x;
  if (idx >= BATCH * HEADS * ENC_T) return;
  const int t  = idx % ENC_T;
  const int bh = idx / ENC_T;
  const int b  = bh / HEADS, h = bh % HEADS;
  float* base = q + ((long)b * ENC_D + h * DH) * ENC_T + t;

  float mx = -INFINITY;
  for (int d = 0; d < DH; ++d) mx = fmaxf(mx, base[(long)d * ENC_T]);
  float s = 0.0f;
  for (int d = 0; d < DH; ++d) s += expf(base[(long)d * ENC_T] - mx);
  const float inv = 1.0f / s;
  for (int d = 0; d < DH; ++d)
    base[(long)d * ENC_T] = expf(base[(long)d * ENC_T] - mx) * inv;
}

// ---------------------------------------------------------------------------
// Softmax over time (512), one wave32 per row; shuffle reductions.
// ---------------------------------------------------------------------------
__global__ void softmax_time(float* __restrict__ k)
{
  const int row  = (blockIdx.x * blockDim.x + threadIdx.x) >> 5;
  const int lane = threadIdx.x & 31;
  if (row >= BATCH * ENC_D) return;
  float* p = k + (long)row * ENC_T;

  float mx = -INFINITY;
  for (int j = lane; j < ENC_T; j += 32) mx = fmaxf(mx, p[j]);
#pragma unroll
  for (int off = 16; off; off >>= 1) mx = fmaxf(mx, __shfl_xor(mx, off, 32));
  float s = 0.0f;
  for (int j = lane; j < ENC_T; j += 32) s += expf(p[j] - mx);
#pragma unroll
  for (int off = 16; off; off >>= 1) s += __shfl_xor(s, off, 32);
  const float inv = 1.0f / s;
  for (int j = lane; j < ENC_T; j += 32) p[j] = expf(p[j] - mx) * inv;
}

// ---------------------------------------------------------------------------
// From proj (B,256,512): means / log_stds outputs, neg_cent B-matrix [r|m*r],
// and cvec[b,j] = nc1+nc4.  Grid: (B, 512), block: 128 (one thread per d).
// ---------------------------------------------------------------------------
__global__ __launch_bounds__(128)
void prep_nc(const float* __restrict__ proj,
             float* __restrict__ Bnc, float* __restrict__ cvec,
             float* __restrict__ means_out, float* __restrict__ logstd_out)
{
  const int b = blockIdx.x, j = blockIdx.y, d = threadIdx.x;
  const long pb = (long)b * PROJ_D * ENC_T;

  const float m    = proj[pb + (long)d * ENC_T + j];
  const float sraw = proj[pb + (long)(FEAT_D + d) * ENC_T + j];
  const float sp = fmaxf(sraw, 0.0f) + log1pf(expf(-fabsf(sraw)));  // softplus
  const float ls = logf(sp);
  const float r  = 1.0f / (sp * sp);     // exp(-2*log_std)

  const long ob = (long)b * FEAT_D * ENC_T + (long)d * ENC_T + j;
  means_out[ob]  = m;
  logstd_out[ob] = ls;

  Bnc[pb + (long)d * ENC_T + j]            = r;
  Bnc[pb + (long)(FEAT_D + d) * ENC_T + j] = m * r;

  __shared__ float red[128];
  red[d] = -ls - 0.5f * m * m * r;
  __syncthreads();
#pragma unroll
  for (int s = 64; s; s >>= 1) {
    if (d < s) red[d] += red[d + s];
    __syncthreads();
  }
  if (d == 0)
    cvec[b * ENC_T + j] = red[0] + 128.0f * (-0.918938533204672742f); // -0.5*log(2*pi)
}

// ---------------------------------------------------------------------------
// Build A_nc (B, 2048, 256): [-0.5*f^2 | f] along K, from features (B,128,2048)
// ---------------------------------------------------------------------------
__global__ void prep_feat(const float* __restrict__ f, float* __restrict__ Anc)
{
  const long n = (long)BATCH * FEAT_T * PROJ_D;
  const long stride = (long)gridDim.x * blockDim.x;
  for (long i = (long)blockIdx.x * blockDim.x + threadIdx.x; i < n; i += stride) {
    const int kk = (int)(i & 255);
    const int t  = (int)((i >> 8) & 2047);
    const int b  = (int)(i >> 19);
    const int d  = kk & 127;
    const float v = f[((long)b * FEAT_D + d) * FEAT_T + t];
    Anc[i] = (kk < FEAT_D) ? (-0.5f * v * v) : v;
  }
}

// ---------------------------------------------------------------------------
// Viterbi forward scan, in place on neg_cent (B,2048,512).
// One WG per batch, 512 threads (one per token j), carry row in LDS.
// ---------------------------------------------------------------------------
__global__ __launch_bounds__(512)
void dp_forward(float* __restrict__ Q,
                const int* __restrict__ elen_, const int* __restrict__ flen_)
{
  const int b = blockIdx.x, j = threadIdx.x;
  const int elen = elen_[b], flen = flen_[b];
  __shared__ float prev[ENC_T];
  prev[j] = 0.0f;
  __syncthreads();

  float* Qb = Q + (long)b * FEAT_T * ENC_T;
  for (int t = 0; t < FEAT_T; ++t) {
    const float rowv = Qb[(long)t * ENC_T + j];
    const float pj   = prev[j];
    const float pjm1 = (j == 0) ? ((t == 0) ? 0.0f : NEGC) : prev[j - 1];
    const float cur  = (j == t) ? NEGC : pj;
    const float val  = rowv + fmaxf(pjm1, cur);
    const int lower  = max(0, elen + t - flen);
    const int upper  = min(elen, t + 1);
    const float out  = (j >= lower && j < upper) ? val : rowv;
    Qb[(long)t * ENC_T + j] = out;
    __syncthreads();
    prev[j] = out;
    __syncthreads();
  }
}

// ---------------------------------------------------------------------------
// Backtrace: one lane per batch. Writes transposed one-hot alignments
// (B, 512, 2048) and integer durations (B, 512).
// ---------------------------------------------------------------------------
__global__ void dp_backward(const float* __restrict__ Q,
                            const int* __restrict__ elen_, const int* __restrict__ flen_,
                            float* __restrict__ align, int* __restrict__ dur)
{
  const int b = threadIdx.x;
  if (b >= BATCH) return;
  const int elen = elen_[b], flen = flen_[b];
  const float* Qb = Q + (long)b * FEAT_T * ENC_T;
  float* Ab = align + (long)b * ENC_T * FEAT_T;
  int* Db = dur + b * ENC_T;

  int tok = elen - 1;
  for (int t = FEAT_T - 1; t >= 0; --t) {
    const bool active = t < flen;
    const int tj = tok & 511;                 // tok mod T_t (python semantics)
    if (active) {
      Ab[(long)tj * FEAT_T + t] = 1.0f;       // transposed alignment
      Db[tj] += 1;
    }
    const float* prow = Qb + (long)((t - 1) & 2047) * ENC_T;   // (t-1) mod T_f
    const float q_cur = prow[tj];
    const float q_m1  = prow[(tok - 1) & 511];
    const bool dec = ((tok != 0) && (tok == t)) || (q_cur < q_m1);
    if (active && dec) tok -= 1;
  }
  Db[elen - 1] += FEAT_T - flen;              // T_f - sum(durations)
}

__global__ void zero_buf(float* __restrict__ p, long n)
{
  const long stride = (long)gridDim.x * blockDim.x;
  for (long i = (long)blockIdx.x * blockDim.x + threadIdx.x; i < n; i += stride)
    p[i] = 0.0f;
}

// ---------------------------------------------------------------------------
// Host-side launcher
// ---------------------------------------------------------------------------
extern "C" void kernel_launch(void* const* d_in, const int* in_sizes, int n_in,
                              void* d_out, int out_size, void* d_ws, size_t ws_size,
                              hipStream_t stream)
{
  (void)in_sizes; (void)n_in; (void)out_size; (void)ws_size;

  const float* encodings = (const float*)d_in[0];
  const int*   enc_len   = (const int*)  d_in[1];
  const float* conditions= (const float*)d_in[2];
  const float* features  = (const float*)d_in[3];
  const int*   feat_len  = (const int*)  d_in[4];
  const float* Wq = (const float*)d_in[5];
  const float* bq = (const float*)d_in[6];
  const float* Wk = (const float*)d_in[7];
  const float* bk = (const float*)d_in[8];
  const float* Wv = (const float*)d_in[9];
  const float* bv = (const float*)d_in[10];
  const float* Wo = (const float*)d_in[11];
  const float* bo = (const float*)d_in[12];
  const float* Wp = (const float*)d_in[13];
  const float* bp = (const float*)d_in[14];

  // ----- output regions (tuple concatenated flat in return order) -----
  float* out_f = (float*)d_out;
  const long N_ALIGN = (long)BATCH * ENC_T * FEAT_T;     // 33,554,432
  const long N_DUR   = (long)BATCH * ENC_T;              // 16,384 (int32)
  const long N_MLS   = (long)BATCH * FEAT_D * ENC_T;     // 2,097,152
  float* out_align  = out_f;
  int*   out_dur    = (int*)(out_f + N_ALIGN);           // int32 bit pattern
  float* out_means  = out_f + N_ALIGN + N_DUR;
  float* out_logstd = out_f + N_ALIGN + N_DUR + N_MLS;

  // ----- workspace layout (floats) -----
  float* ws = (float*)d_ws;
  const long SZ_QKV = (long)BATCH * ENC_D * ENC_T;       // 3,145,728
  float* w_q    = ws;                         // also reused for enc' (enc2)
  float* w_k    = w_q  + SZ_QKV;
  float* w_v    = w_k  + SZ_QKV;
  float* w_ctx  = w_v  + SZ_QKV;              // (B,H,96,96)
  float* w_att  = w_ctx + (long)BATCH * HEADS * DH * DH;
  float* w_proj = w_att + SZ_QKV;             // (B,256,512)
  float* w_bnc  = w_proj + (long)BATCH * PROJ_D * ENC_T;
  float* w_cvec = w_bnc  + (long)BATCH * PROJ_D * ENC_T;
  float* w_anc  = w_cvec + (long)BATCH * ENC_T;           // (B,2048,256)
  float* w_nc   = w_anc  + (long)BATCH * FEAT_T * PROJ_D; // (B,2048,512)
  float* w_enc2 = w_q;   // reuse: q is dead after the att GEMM

  const dim3 blk128(128), blk256(256), blk512(512);

  // 0) zero alignments + durations output regions
  zero_buf<<<1024, blk256, 0, stream>>>(out_align, N_ALIGN + N_DUR);

  // 1-3) QKV projections: out = W * X + b   (M=192, N=512, K=192, 32 batches)
  {
    const dim3 g(ENC_T / TILE_N, ENC_D / TILE_M, BATCH);
    gemm_wmma_f16<<<g, blk128, 0, stream>>>(Wq, 0, ENC_D, 0,
                                            encodings, (long)ENC_D * ENC_T, ENC_T, 0,
                                            w_q, (long)ENC_D * ENC_T, ENC_T,
                                            ENC_D, ENC_T, ENC_D,
                                            bq, nullptr, 0, 0, nullptr, nullptr, nullptr, MODE_BIAS);
    gemm_wmma_f16<<<g, blk128, 0, stream>>>(Wk, 0, ENC_D, 0,
                                            conditions, (long)ENC_D * ENC_T, ENC_T, 0,
                                            w_k, (long)ENC_D * ENC_T, ENC_T,
                                            ENC_D, ENC_T, ENC_D,
                                            bk, nullptr, 0, 0, nullptr, nullptr, nullptr, MODE_BIAS);
    gemm_wmma_f16<<<g, blk128, 0, stream>>>(Wv, 0, ENC_D, 0,
                                            conditions, (long)ENC_D * ENC_T, ENC_T, 0,
                                            w_v, (long)ENC_D * ENC_T, ENC_T,
                                            ENC_D, ENC_T, ENC_D,
                                            bv, nullptr, 0, 0, nullptr, nullptr, nullptr, MODE_BIAS);
  }

  // 4) softmaxes: q over channel (d), k over time (t)
  softmax_channel<<<(BATCH * HEADS * ENC_T + 255) / 256, blk256, 0, stream>>>(w_q);
  softmax_time<<<(BATCH * ENC_D * 32 + 255) / 256, blk256, 0, stream>>>(w_k);

  // 5) ctx[bh] = k_s (96x512) * v^T (512x96)    [transB]
  {
    const dim3 g((DH + TILE_N - 1) / TILE_N, (DH + TILE_M - 1) / TILE_M, BATCH * HEADS);
    gemm_wmma_f16<<<g, blk128, 0, stream>>>(w_k, (long)DH * ENC_T, ENC_T, 0,
                                            w_v, (long)DH * ENC_T, ENC_T, 1,
                                            w_ctx, (long)DH * DH, DH,
                                            DH, DH, ENC_T,
                                            nullptr, nullptr, 0, 0, nullptr, nullptr, nullptr, 0);
  }

  // 6) att[bh] = ctx^T (96x96) * q_s (96x512)   [transA]
  {
    const dim3 g(ENC_T / TILE_N, (DH + TILE_M - 1) / TILE_M, BATCH * HEADS);
    gemm_wmma_f16<<<g, blk128, 0, stream>>>(w_ctx, (long)DH * DH, DH, 1,
                                            w_q, (long)DH * ENC_T, ENC_T, 0,
                                            w_att, (long)DH * ENC_T, ENC_T,
                                            DH, ENC_T, DH,
                                            nullptr, nullptr, 0, 0, nullptr, nullptr, nullptr, 0);
  }

  // 7) enc' = Wo * att + bo + encodings (residual); into reused q buffer
  {
    const dim3 g(ENC_T / TILE_N, ENC_D / TILE_M, BATCH);
    gemm_wmma_f16<<<g, blk128, 0, stream>>>(Wo, 0, ENC_D, 0,
                                            w_att, (long)ENC_D * ENC_T, ENC_T, 0,
                                            w_enc2, (long)ENC_D * ENC_T, ENC_T,
                                            ENC_D, ENC_T, ENC_D,
                                            bo, encodings, (long)ENC_D * ENC_T, ENC_T,
                                            nullptr, nullptr, nullptr, MODE_BIAS | MODE_RESID);
  }

  // 8) proj = (Wp * enc' + bp) masked by enc_len columns   (M=256)
  {
    const dim3 g(ENC_T / TILE_N, PROJ_D / TILE_M, BATCH);
    gemm_wmma_f16<<<g, blk128, 0, stream>>>(Wp, 0, ENC_D, 0,
                                            w_enc2, (long)ENC_D * ENC_T, ENC_T, 0,
                                            w_proj, (long)PROJ_D * ENC_T, ENC_T,
                                            PROJ_D, ENC_T, ENC_D,
                                            bp, nullptr, 0, 0, nullptr,
                                            enc_len, nullptr, MODE_BIAS | MODE_COLMASK);
  }

  // 9) neg_cent B-matrix [r | m*r], cvec, means & log_stds outputs
  {
    const dim3 g(BATCH, ENC_T);
    prep_nc<<<g, blk128, 0, stream>>>(w_proj, w_bnc, w_cvec, out_means, out_logstd);
  }

  // 10) neg_cent A-matrix [-0.5*f^2 | f]
  prep_feat<<<2048, blk256, 0, stream>>>(features, w_anc);

  // 11) neg_cent = A_nc (2048x256) * B_nc (256x512) + cvec, band-masked
  {
    const dim3 g(ENC_T / TILE_N, FEAT_T / TILE_M, BATCH);
    gemm_wmma_f16<<<g, blk128, 0, stream>>>(w_anc, (long)FEAT_T * PROJ_D, PROJ_D, 0,
                                            w_bnc, (long)PROJ_D * ENC_T, ENC_T, 0,
                                            w_nc, (long)FEAT_T * ENC_T, ENC_T,
                                            FEAT_T, ENC_T, PROJ_D,
                                            nullptr, nullptr, 0, 0,
                                            w_cvec, enc_len, feat_len, MODE_NEGCENT);
  }

  // 12) Viterbi forward scan (in place on neg_cent)
  dp_forward<<<BATCH, blk512, 0, stream>>>(w_nc, enc_len, feat_len);

  // 13) backtrace -> transposed alignments + durations
  dp_backward<<<1, dim3(32), 0, stream>>>(w_nc, enc_len, feat_len, out_align, out_dur);
}
